// SwinWindowBlock_63728724738258
// MI455X (gfx1250) — matrix operand order repaired and checked by reading, not compiled, not run on hardware
//
#include <hip/hip_runtime.h>
#include <hip/hip_bf16.h>
#include <math.h>

// ---------------------------------------------------------------------------
// Swin window block, fused per-window kernel for gfx1250 (MI455X).
// One workgroup (8 wave32) per 64x384 window. All GEMMs via
// v_wmma_f32_16x16x32_f16 with f32 accumulation. Weight GEMMs are register-
// blocked 2 M-tiles x 2 N-tiles per wave: each group issues 4 independent
// wmmas from 2 A fragments (LDS) + 2 B fragments (global f16 weights),
// halving B-operand bandwidth vs 1Mx4N and balancing ds/vmem issue.
// Weights pre-converted to f16 into d_ws (2.36 MB, L2-resident at 192 MB).
// ---------------------------------------------------------------------------

typedef _Float16 h16;
typedef __attribute__((ext_vector_type(16))) _Float16 v16h;
typedef __attribute__((ext_vector_type(8)))  float    v8f;

#define DIMC   384
#define SEQ    64
#define NHEADS 8
#define HD     48
#define HIDN   768
#define QKVN   1152

union HFrag { v16h v; h16 e[16]; };

#define WMMA_F16(a, b, c)                                                     \
  __builtin_amdgcn_wmma_f32_16x16x32_f16(false, (a), false, (b), (short)0,    \
                                         (c), false, false)

// A fragment (16x32 f16) from LDS, row-major activations [M][lda].
// Contiguous k order (same as load_b_*): lanes 0-15 -> M rows, K(e) = k0 + e + 16*(lane>=16).
__device__ __forceinline__ v16h load_a_lds(const h16* A, int lda, int m0, int k0,
                                           int klim, int lane) {
  HFrag f;
  const h16* row = A + (size_t)(m0 + (lane & 15)) * lda;
  const int kb = (lane & 16) ? 16 : 0;
#pragma unroll
  for (int e = 0; e < 16; ++e) {
    int k = k0 + kb + e;
    f.e[e] = (k < klim) ? row[k] : (h16)0.0f;
  }
  return f.v;
}

// B fragment (32x16 f16) where B[k][n] = W[n][k], W row-major [N][ldk]
// (PyTorch Linear convention). K(e) = k0 + e + 16*(lane>=16), N = n0+lane&15.
__device__ __forceinline__ v16h load_b_nk(const h16* W, int ldk, int n0, int k0,
                                          int klim, int lane) {
  HFrag f;
  const h16* row = W + (size_t)(n0 + (lane & 15)) * ldk;
  const int kb = k0 + ((lane & 16) ? 16 : 0);
#pragma unroll
  for (int e = 0; e < 16; ++e) {
    int k = kb + e;
    f.e[e] = (k < klim) ? row[k] : (h16)0.0f;
  }
  return f.v;
}

// B fragment (32x16 f16) from row-major-K storage B[k][n] with stride ldn
// (used for att @ V where V rows are tokens).
__device__ __forceinline__ v16h load_b_kn(const h16* B, int ldn, int n0, int k0,
                                          int lane) {
  HFrag f;
  const int n  = n0 + (lane & 15);
  const int kb = k0 + ((lane & 16) ? 16 : 0);
#pragma unroll
  for (int e = 0; e < 16; ++e)
    f.e[e] = B[(size_t)(kb + e) * ldn + n];
  return f.v;
}

// C/D epilogue: lane/VGPR -> (row = m0 + r + 8*(lane>=16), col = n0 + lane&15)
// MODE 0: +bias, store f16. MODE 1: +bias, exact GELU, store f16.
// MODE 2: +bias, residual add into f32 R.
template <int MODE>
__device__ __forceinline__ void epilogue(v8f acc, int m0, int n0, int lane,
                                         const float* bias, h16* outH, int ldo,
                                         float* Rres) {
  const int col = n0 + (lane & 15);
  const float bv = bias[col];
  const int rbase = m0 + ((lane & 16) ? 8 : 0);
#pragma unroll
  for (int r = 0; r < 8; ++r) {
    const int row = rbase + r;
    float v = acc[r] + bv;
    if (MODE == 0) {
      outH[(size_t)row * ldo + col] = (h16)v;
    } else if (MODE == 1) {
      v = 0.5f * v * (1.0f + erff(v * 0.70710678118f));
      outH[(size_t)row * ldo + col] = (h16)v;
    } else {
      float* p = Rres + (size_t)row * DIMC + col;
      *p = *p + v;
    }
  }
}

// Out[64xN] = A[64xK] * W^T + bias. Wave w owns M-pair (w&1: rows 0-31/32-63)
// and N-quarter (w>>1). Groups of 2 M-tiles x 2 N-tiles: 2 A + 2 B fragment
// loads feed 4 wmmas. Requires (N/16/4) % 2 == 0 (N = 1152, 768, 384: ok).
template <int MODE>
__device__ __forceinline__ void gemm_wf16(const h16* Alds, int lda, const h16* W,
                                          int K, int N, const float* bias,
                                          h16* outH, int ldo, float* Rres, int tid) {
  const int wave = tid >> 5, lane = tid & 31;
  const int m0 = (wave & 1) << 5;        // 0 or 32; tiles m0, m0+16
  const int quarter = wave >> 1;         // 0..3
  const int nq = N >> 6;                 // N-tiles per quarter
  for (int j0 = quarter * nq; j0 < quarter * nq + nq; j0 += 2) {
    v8f acc00 = {}, acc01 = {}, acc10 = {}, acc11 = {};
    for (int k0 = 0; k0 < K; k0 += 32) {
      v16h a0 = load_a_lds(Alds, lda, m0, k0, K, lane);
      v16h a1 = load_a_lds(Alds, lda, m0 + 16, k0, K, lane);
      v16h b0 = load_b_nk(W, K, (j0 + 0) << 4, k0, K, lane);
      v16h b1 = load_b_nk(W, K, (j0 + 1) << 4, k0, K, lane);
      acc00 = WMMA_F16(a0, b0, acc00);
      acc01 = WMMA_F16(a0, b1, acc01);
      acc10 = WMMA_F16(a1, b0, acc10);
      acc11 = WMMA_F16(a1, b1, acc11);
    }
    epilogue<MODE>(acc00, m0,      (j0 + 0) << 4, lane, bias, outH, ldo, Rres);
    epilogue<MODE>(acc01, m0,      (j0 + 1) << 4, lane, bias, outH, ldo, Rres);
    epilogue<MODE>(acc10, m0 + 16, (j0 + 0) << 4, lane, bias, outH, ldo, Rres);
    epilogue<MODE>(acc11, m0 + 16, (j0 + 1) << 4, lane, bias, outH, ldo, Rres);
  }
}

// LayerNorm over C=384 of f32 R -> f16 Y. 4 threads per row (96 cols each).
__device__ __forceinline__ void layernorm_lds(const float* R, h16* Y,
                                              const float* w, const float* b,
                                              float* RED, int tid) {
  const int row = tid >> 2, q = tid & 3;
  const float* x = R + (size_t)row * DIMC + q * 96;
  float s = 0.f, ss = 0.f;
#pragma unroll 4
  for (int i = 0; i < 96; ++i) { float v = x[i]; s += v; ss += v * v; }
  RED[(q * 64 + row) * 2 + 0] = s;
  RED[(q * 64 + row) * 2 + 1] = ss;
  __syncthreads();
  float S0 = 0.f, S1 = 0.f;
#pragma unroll
  for (int qq = 0; qq < 4; ++qq) {
    S0 += RED[(qq * 64 + row) * 2 + 0];
    S1 += RED[(qq * 64 + row) * 2 + 1];
  }
  const float mu  = S0 * (1.0f / DIMC);
  const float var = S1 * (1.0f / DIMC) - mu * mu;
  const float rs  = rsqrtf(var + 1e-5f);
  h16* y = Y + (size_t)row * DIMC + q * 96;
  const float* wv = w + q * 96;
  const float* bv = b + q * 96;
#pragma unroll 4
  for (int i = 0; i < 96; ++i) y[i] = (h16)((x[i] - mu) * rs * wv[i] + bv[i]);
  __syncthreads();
}

// LDS partition (bytes):
#define OFF_R    0                               // f32 residual  64*384*4 = 98304
#define OFF_Y    (OFF_R + SEQ * DIMC * 4)        // f16 LN/act    64*384*2 = 49152
#define OFF_S    (OFF_Y + SEQ * DIMC * 2)        // f16 qkv/mlp   64*1152*2 = 147456
#define OFF_ATT  (OFF_S + SEQ * QKVN * 2)        // f32 scores    64*64*4 = 16384
#define OFF_A16  (OFF_ATT + SEQ * SEQ * 4)       // f16 probs     64*64*2 = 8192
#define OFF_RED  (OFF_A16 + SEQ * SEQ * 2)       // f32 reduce    512*4  = 2048
#define SMEM_BYTES (OFF_RED + 512 * 4)           // = 321536 <= 320KB WGP LDS

__global__ __launch_bounds__(256, 1)
void swin_window_block_kernel(const float* __restrict__ x,
                              const float* __restrict__ n1w, const float* __restrict__ n1b,
                              const float* __restrict__ in_b, const float* __restrict__ out_b,
                              const float* __restrict__ n2w, const float* __restrict__ n2b,
                              const float* __restrict__ b1,  const float* __restrict__ b2,
                              const h16* __restrict__ Wqkv, const h16* __restrict__ Wout,
                              const h16* __restrict__ W1,   const h16* __restrict__ W2,
                              float* __restrict__ out) {
  extern __shared__ char smem[];
  float* R    = (float*)(smem + OFF_R);
  h16*   Y    = (h16*)  (smem + OFF_Y);
  h16*   S    = (h16*)  (smem + OFF_S);
  float* ATT  = (float*)(smem + OFF_ATT);
  h16*   A16  = (h16*)  (smem + OFF_A16);
  float* RED  = (float*)(smem + OFF_RED);

  const int tid  = threadIdx.x;
  const int wave = tid >> 5, lane = tid & 31;
  const int wid  = blockIdx.x;              // 0..2047
  const int bimg = wid >> 8;
  const int rem  = wid & 255;
  const int wh   = rem >> 4, ww = rem & 15;
  const float* xb = x + (size_t)bimg * DIMC * 128 * 128;

  // ---- load window -> f32 residual buffer R [64 tokens][384 ch] ----
  for (int i = tid; i < SEQ * DIMC; i += 256) {
    const int c = i >> 6, s = i & 63;
    const int ty = s >> 3, tx = s & 7;
    R[(size_t)s * DIMC + c] =
        xb[((size_t)c * 128 + (wh * 8 + ty)) * 128 + (ww * 8 + tx)];
  }
  __syncthreads();

  // ---- LN1 -> Y(f16) ----
  layernorm_lds(R, Y, n1w, n1b, RED, tid);

  // ---- QKV: S[64x1152] = Y * Wqkv^T + in_b ----
  gemm_wf16<0>(Y, DIMC, Wqkv, DIMC, QKVN, in_b, S, QKVN, nullptr, tid);
  __syncthreads();

  // ---- attention, per head ----
  const float scale = 0.14433756729740645f;  // 1/sqrt(48)
  for (int h = 0; h < NHEADS; ++h) {
    const int qoff = h * HD, koff = DIMC + h * HD, voff = 2 * DIMC + h * HD;
    // scores: [64x64] = Q_h * K_h^T   (K dim = 48, zero-padded tail)
    for (int t = wave; t < 16; t += 8) {
      const int m0 = (t & 3) << 4, n0 = (t >> 2) << 4;
      v8f acc = {};
#pragma unroll
      for (int k0 = 0; k0 < 64; k0 += 32) {
        v16h a = load_a_lds(S + qoff, QKVN, m0, k0, HD, lane);
        v16h b = load_b_nk(S + koff, QKVN, n0, k0, HD, lane);
        acc = WMMA_F16(a, b, acc);
      }
      const int col = n0 + (lane & 15);
      const int rbase = m0 + ((lane & 16) ? 8 : 0);
#pragma unroll
      for (int r = 0; r < 8; ++r) ATT[(size_t)(rbase + r) * SEQ + col] = acc[r];
    }
    __syncthreads();
    // softmax rows (scale folded in)
    if (tid < SEQ) {
      float* arow = ATT + (size_t)tid * SEQ;
      float mx = -3.4e38f;
      for (int j = 0; j < SEQ; ++j) mx = fmaxf(mx, arow[j]);
      float sum = 0.f;
      for (int j = 0; j < SEQ; ++j) {
        float e = __expf(scale * (arow[j] - mx));
        arow[j] = e;
        sum += e;
      }
      const float inv = 1.0f / sum;
      h16* prow = A16 + (size_t)tid * SEQ;
      for (int j = 0; j < SEQ; ++j) prow[j] = (h16)(arow[j] * inv);
    }
    __syncthreads();
    // O_h[64x48] = att * V_h  -> Y columns [h*48, h*48+48)
    for (int t = wave; t < 12; t += 8) {
      const int m0 = (t & 3) << 4, n0 = (t >> 2) << 4;
      v8f acc = {};
#pragma unroll
      for (int k0 = 0; k0 < SEQ; k0 += 32) {
        v16h a = load_a_lds(A16, SEQ, m0, k0, SEQ, lane);
        v16h b = load_b_kn(S + voff, QKVN, n0, k0, lane);
        acc = WMMA_F16(a, b, acc);
      }
      const int col = qoff + n0 + (lane & 15);
      const int rbase = m0 + ((lane & 16) ? 8 : 0);
#pragma unroll
      for (int r = 0; r < 8; ++r) Y[(size_t)(rbase + r) * DIMC + col] = (h16)acc[r];
    }
    __syncthreads();
  }

  // ---- out projection + residual: R += Y * Wout^T + out_b ----
  gemm_wf16<2>(Y, DIMC, Wout, DIMC, DIMC, out_b, nullptr, 0, R, tid);
  __syncthreads();

  // ---- LN2 -> Y, MLP ----
  layernorm_lds(R, Y, n2w, n2b, RED, tid);
  gemm_wf16<1>(Y, DIMC, W1, DIMC, HIDN, b1, S, HIDN, nullptr, tid);  // gelu -> S
  __syncthreads();
  gemm_wf16<2>(S, HIDN, W2, HIDN, DIMC, b2, nullptr, 0, R, tid);     // R += mlp
  __syncthreads();

  // ---- window reverse store ----
  for (int i = tid; i < SEQ * DIMC; i += 256) {
    const int c = i >> 6, s = i & 63;
    const int ty = s >> 3, tx = s & 7;
    out[((size_t)bimg * DIMC + c) * 16384 + (size_t)(wh * 8 + ty) * 128 +
        (ww * 8 + tx)] = R[(size_t)s * DIMC + c];
  }
}

// ---- weight f32 -> f16 pre-pass ----
__global__ void cvt_f32_to_f16_kernel(const float* __restrict__ a,
                                      h16* __restrict__ o, int n) {
  int i = blockIdx.x * 256 + threadIdx.x;
  if (i < n) o[i] = (h16)a[i];
}

extern "C" void kernel_launch(void* const* d_in, const int* in_sizes, int n_in,
                              void* d_out, int out_size, void* d_ws, size_t ws_size,
                              hipStream_t stream) {
  const float* x     = (const float*)d_in[0];
  const float* n1w   = (const float*)d_in[1];
  const float* n1b   = (const float*)d_in[2];
  const float* in_w  = (const float*)d_in[3];   // [1152,384]
  const float* in_b  = (const float*)d_in[4];
  const float* out_w = (const float*)d_in[5];   // [384,384]
  const float* out_b = (const float*)d_in[6];
  const float* n2w   = (const float*)d_in[7];
  const float* n2b   = (const float*)d_in[8];
  const float* w1    = (const float*)d_in[9];   // [768,384]
  const float* b1    = (const float*)d_in[10];
  const float* w2    = (const float*)d_in[11];  // [384,768]
  const float* b2    = (const float*)d_in[12];
  float* out = (float*)d_out;

  // f16 weight copies in workspace
  h16* Wqkv = (h16*)d_ws;                       // 1152*384
  h16* Wout = Wqkv + QKVN * DIMC;               // 384*384
  h16* W1   = Wout + DIMC * DIMC;               // 768*384
  h16* W2   = W1 + HIDN * DIMC;                 // 384*768
  (void)ws_size; (void)n_in; (void)in_sizes; (void)out_size;

  const int nqkv = QKVN * DIMC, nout = DIMC * DIMC, n1 = HIDN * DIMC, n2 = DIMC * HIDN;
  cvt_f32_to_f16_kernel<<<(nqkv + 255) / 256, 256, 0, stream>>>(in_w, Wqkv, nqkv);
  cvt_f32_to_f16_kernel<<<(nout + 255) / 256, 256, 0, stream>>>(out_w, Wout, nout);
  cvt_f32_to_f16_kernel<<<(n1 + 255) / 256, 256, 0, stream>>>(w1, W1, n1);
  cvt_f32_to_f16_kernel<<<(n2 + 255) / 256, 256, 0, stream>>>(w2, W2, n2);

  (void)hipFuncSetAttribute((const void*)swin_window_block_kernel,
                            hipFuncAttributeMaxDynamicSharedMemorySize, SMEM_BYTES);

  swin_window_block_kernel<<<2048, 256, SMEM_BYTES, stream>>>(
      x, n1w, n1b, in_b, out_b, n2w, n2b, b1, b2, Wqkv, Wout, W1, W2, out);
}
